// MSVED_36326833389682
// MI455X (gfx1250) — matrix operand order repaired and checked by reading, not compiled, not run on hardware
//
#include <hip/hip_runtime.h>
#include <hip/hip_bf16.h>
#include <math.h>

// ---------------------------------------------------------------------------
// MSVED VAE forward on gfx1250 (wave32, WMMA).
//  - Large GEMMs: bf16 WMMA 16x16x32, f32 accum, 2Mx2N register blocking per
//    wave (2 b128 loads per WMMA), global_prefetch for the HBM-streamed B.
//  - GRU recurrences: persistent kernels, one grid-sync per step; h staged to
//    LDS via GLOBAL_LOAD_ASYNC_TO_LDS_B128 (ASYNCcnt) each step, float4 dots.
//  - bf16 staging kernels are 2D (no integer division) -> pure bandwidth.
// ---------------------------------------------------------------------------

typedef __attribute__((ext_vector_type(16))) __bf16 bf16x16;
typedef __attribute__((ext_vector_type(8)))  __bf16 bf16x8;
typedef __attribute__((ext_vector_type(8)))  float  f32x8;

static constexpr int H    = 1024;
static constexpr int Z    = 256;
static constexpr int V    = 10000;
static constexpr int T    = 512;
static constexpr int LAB  = 64;
static constexpr int G3H  = 3 * H;          // 3072
static constexpr int VP   = 10016;          // V padded to multiple of 32 (K dim)
static constexpr int NPAD = 10016;          // V padded to multiple of 32 (N dim)

// ---------------------------------------------------------------------------
// WMMA helpers (CDNA5 16x16x32 bf16 -> f32). Layouts per ISA 7.12.2.
// ---------------------------------------------------------------------------
__device__ inline f32x8 wmma_bf16(bf16x16 a, bf16x16 b, f32x8 c) {
  return __builtin_amdgcn_wmma_f32_16x16x32_bf16(
      false, a, false, b, (short)0, c, false, false);
}

// A fragment, 16x32 bf16, row-major source. Lane l: m = l&15, g = l>>4.
__device__ inline bf16x16 load_a_frag(const __bf16* row, int k0, int g) {
  bf16x8 lo = *(const bf16x8*)(row + k0 + g * 8);
  bf16x8 hi = *(const bf16x8*)(row + k0 + 16 + g * 8);
  return __builtin_shufflevector(lo, hi, 0,1,2,3,4,5,6,7,8,9,10,11,12,13,14,15);
}

// B fragment, 32x16 bf16 with B[k][n] = W[n][k] (W row-major).
__device__ inline bf16x16 load_b_frag(const __bf16* row, int k0, int g) {
  bf16x8 lo = *(const bf16x8*)(row + k0 + g * 8);
  bf16x8 hi = *(const bf16x8*)(row + k0 + 16 + g * 8);
  return __builtin_shufflevector(lo, hi, 0,1,2,3,4,5,6,7,8,9,10,11,12,13,14,15);
}

// ---------------------------------------------------------------------------
// Grid-wide split barrier (device-scope atomics), for persistent recurrences.
// ---------------------------------------------------------------------------
__device__ inline void grid_sync(int* bar, int nblk) {
  __syncthreads();
  if (threadIdx.x == 0) {
    __threadfence();
    int gen = __hip_atomic_load(&bar[1], __ATOMIC_RELAXED, __HIP_MEMORY_SCOPE_AGENT);
    if (__hip_atomic_fetch_add(&bar[0], 1, __ATOMIC_ACQ_REL, __HIP_MEMORY_SCOPE_AGENT)
        == nblk - 1) {
      __hip_atomic_store(&bar[0], 0, __ATOMIC_RELAXED, __HIP_MEMORY_SCOPE_AGENT);
      __hip_atomic_fetch_add(&bar[1], 1, __ATOMIC_RELEASE, __HIP_MEMORY_SCOPE_AGENT);
    } else {
      while (__hip_atomic_load(&bar[1], __ATOMIC_ACQUIRE, __HIP_MEMORY_SCOPE_AGENT)
             == gen) {
        __builtin_amdgcn_s_sleep(2);
      }
    }
  }
  __syncthreads();
  __threadfence();  // acquire for every thread's subsequent loads
}

// Async copy of h (1024 floats) into LDS: one 16B ASYNC transfer per thread,
// then drain ASYNCcnt and barrier. LDS byte address = low 32 bits of the
// generic __shared__ pointer (flat->LDS truncation, ISA 10.2).
__device__ inline void stage_h_to_lds(float* sh, const float* hold) {
  unsigned lds_off = (unsigned)(uintptr_t)(sh + threadIdx.x * 4);
  const float* g = hold + threadIdx.x * 4;
  asm volatile("global_load_async_to_lds_b128 %0, %1, off"
               :: "v"(lds_off), "v"(g) : "memory");
  asm volatile("s_wait_asynccnt 0x0" ::: "memory");
  __syncthreads();
}

// ---------------------------------------------------------------------------
// f32 -> bf16 staging, 2D grids (no integer division; coalesced rows)
// ---------------------------------------------------------------------------
__global__ void cvt_pad_cols(const float* __restrict__ src, __bf16* __restrict__ dst,
                             int C, int Cp) {
  int r = blockIdx.y;
  int c = blockIdx.x * blockDim.x + threadIdx.x;
  if (c < Cp)
    dst[(size_t)r * Cp + c] = (c < C) ? (__bf16)src[(size_t)r * C + c] : (__bf16)0.0f;
}

__global__ void cvt_pad_rows(const float* __restrict__ src, __bf16* __restrict__ dst,
                             int R, int C) {
  int r = blockIdx.y;
  int c = blockIdx.x * blockDim.x + threadIdx.x;
  size_t i = (size_t)r * C + c;
  dst[i] = (r < R) ? (__bf16)src[i] : (__bf16)0.0f;
}

__global__ void init_state(float* henc, float* hdec, int* bar) {
  for (int i = threadIdx.x; i < 4096; i += 256) henc[i] = 0.0f;
  for (int i = threadIdx.x; i < 2048; i += 256) hdec[i] = 0.0f;
  if (threadIdx.x < 64) bar[threadIdx.x] = 0;
}

// ---------------------------------------------------------------------------
// Encoder input projections, 2Mx2N register blocking per wave.
// M=512 (16 tile-pairs), N=3072 (96 tile-pairs), K=VP.
// 1536 tile-groups/dir * 2 dirs = 3072 waves = 384 blocks x 8 waves.
// ---------------------------------------------------------------------------
__global__ void __launch_bounds__(256) gemm_enc_proj(
    const __bf16* __restrict__ xbf,
    const __bf16* __restrict__ wf, const __bf16* __restrict__ wb,
    const float* __restrict__ bih_f, const float* __restrict__ bih_b,
    float* __restrict__ gif, float* __restrict__ gib) {
  int wid  = blockIdx.x * 8 + (threadIdx.x >> 5);
  int lane = threadIdx.x & 31;
  int dir  = (wid >= 1536) ? 1 : 0;
  wid -= dir * 1536;
  int nt2 = wid % 96, mt2 = wid / 96;   // n-fastest: waves in a block share A
  const __bf16* W    = dir ? wb : wf;
  const float*  bias = dir ? bih_b : bih_f;
  float*        out  = dir ? gib : gif;
  int g = lane >> 4, q = lane & 15;
  int m0 = mt2 * 32, n0 = nt2 * 32;
  const __bf16* ar0 = xbf + (size_t)(m0 + q) * VP;
  const __bf16* ar1 = xbf + (size_t)(m0 + 16 + q) * VP;
  const __bf16* br0 = W + (size_t)(n0 + q) * VP;
  const __bf16* br1 = W + (size_t)(n0 + 16 + q) * VP;
  f32x8 acc00 = {0,0,0,0,0,0,0,0}, acc01 = acc00, acc10 = acc00, acc11 = acc00;
  for (int k0 = 0; k0 < VP; k0 += 32) {
    // hide first-touch HBM latency for the streamed B rows (speculative,
    // silently dropped past the buffer end) -> global_prefetch_b8
    __builtin_prefetch(br0 + k0 + 1024, 0, 0);
    __builtin_prefetch(br1 + k0 + 1024, 0, 0);
    bf16x16 a0 = load_a_frag(ar0, k0, g);
    bf16x16 a1 = load_a_frag(ar1, k0, g);
    bf16x16 b0 = load_b_frag(br0, k0, g);
    bf16x16 b1 = load_b_frag(br1, k0, g);
    acc00 = wmma_bf16(a0, b0, acc00);
    acc01 = wmma_bf16(a0, b1, acc01);
    acc10 = wmma_bf16(a1, b0, acc10);
    acc11 = wmma_bf16(a1, b1, acc11);
  }
  float bv0 = bias[n0 + q], bv1 = bias[n0 + 16 + q];
#pragma unroll
  for (int r = 0; r < 8; ++r) {
    int ma = m0 + g * 8 + r, mb = m0 + 16 + g * 8 + r;
    out[(size_t)ma * G3H + n0 + q]      = acc00[r] + bv0;
    out[(size_t)ma * G3H + n0 + 16 + q] = acc01[r] + bv1;
    out[(size_t)mb * G3H + n0 + q]      = acc10[r] + bv0;
    out[(size_t)mb * G3H + n0 + 16 + q] = acc11[r] + bv1;
  }
}

// ---------------------------------------------------------------------------
// Output projection, 2Mx2N blocking. M=512 (16 pairs), N=NPAD (313 pairs),
// K=1024. 5008 waves = 626 blocks x 8 waves.
// ---------------------------------------------------------------------------
__global__ void __launch_bounds__(256) gemm_out_proj(
    const __bf16* __restrict__ hsbf, const __bf16* __restrict__ fcbf,
    const float* __restrict__ fc_b, float* __restrict__ out) {
  int wid  = blockIdx.x * 8 + (threadIdx.x >> 5);
  int lane = threadIdx.x & 31;
  int nt2 = wid % 313, mt2 = wid / 313;
  int g = lane >> 4, q = lane & 15;
  int m0 = mt2 * 32, n0 = nt2 * 32;
  const __bf16* ar0 = hsbf + (size_t)(m0 + q) * H;
  const __bf16* ar1 = hsbf + (size_t)(m0 + 16 + q) * H;
  const __bf16* br0 = fcbf + (size_t)(n0 + q) * H;
  const __bf16* br1 = fcbf + (size_t)(n0 + 16 + q) * H;
  f32x8 acc00 = {0,0,0,0,0,0,0,0}, acc01 = acc00, acc10 = acc00, acc11 = acc00;
  for (int k0 = 0; k0 < H; k0 += 32) {
    bf16x16 a0 = load_a_frag(ar0, k0, g);
    bf16x16 a1 = load_a_frag(ar1, k0, g);
    bf16x16 b0 = load_b_frag(br0, k0, g);
    bf16x16 b1 = load_b_frag(br1, k0, g);
    acc00 = wmma_bf16(a0, b0, acc00);
    acc01 = wmma_bf16(a0, b1, acc01);
    acc10 = wmma_bf16(a1, b0, acc10);
    acc11 = wmma_bf16(a1, b1, acc11);
  }
  int na = n0 + q, nb = n0 + 16 + q;
  float bv0 = (na < V) ? fc_b[na] : 0.f;
  float bv1 = (nb < V) ? fc_b[nb] : 0.f;
#pragma unroll
  for (int r = 0; r < 8; ++r) {
    int ma = m0 + g * 8 + r, mb = m0 + 16 + g * 8 + r;
    if (na < V) {
      out[(size_t)ma * V + na] = acc00[r] + bv0;
      out[(size_t)mb * V + na] = acc10[r] + bv0;
    }
    if (nb < V) {
      out[(size_t)ma * V + nb] = acc01[r] + bv1;
      out[(size_t)mb * V + nb] = acc11[r] + bv1;
    }
  }
}

// ---------------------------------------------------------------------------
// GRU step body: block owns 32 hidden channels, computes 96 gate dot products
// (float4 loads, h from LDS), then the gate math.
// ---------------------------------------------------------------------------
__device__ inline void gru_rows(const float* __restrict__ Whh,
                                const float* __restrict__ bhh,
                                const float* __restrict__ sh_h,
                                float* __restrict__ gh,
                                int j0, int lane, int wave) {
  const float4* h4 = (const float4*)sh_h;
  for (int rr = wave * 12; rr < wave * 12 + 12; ++rr) {
    int gate = rr >> 5, jj = rr & 31;
    int row  = gate * H + j0 + jj;
    const float4* w4 = (const float4*)(Whh + (size_t)row * H);
    float acc = 0.f;
#pragma unroll
    for (int i = 0; i < 8; ++i) {
      float4 hv = h4[lane + 32 * i];
      float4 wv = w4[lane + 32 * i];
      acc += hv.x * wv.x + hv.y * wv.y + hv.z * wv.z + hv.w * wv.w;
    }
    for (int m = 16; m; m >>= 1) acc += __shfl_xor(acc, m, 32);
    if (lane == 0) gh[rr] = acc + bhh[row];
  }
}

// Encoder GRU recurrences (fwd + bwd concurrently), 64 persistent blocks.
__global__ void __launch_bounds__(256, 1) enc_gru(
    const float* __restrict__ wf_Whh, const float* __restrict__ wf_bhh,
    const float* __restrict__ wb_Whh, const float* __restrict__ wb_bhh,
    const float* __restrict__ gif, const float* __restrict__ gib,
    float* __restrict__ henc, int* bar) {
  const int NB = 32;
  int dir = (blockIdx.x >= NB) ? 1 : 0;
  int b   = blockIdx.x & (NB - 1);
  const float* Whh = dir ? wb_Whh : wf_Whh;
  const float* bhh = dir ? wb_bhh : wf_bhh;
  const float* gi  = dir ? gib : gif;
  float* hbase = henc + dir * 2048;           // [2][1024]
  int j0   = b * 32;
  int lane = threadIdx.x & 31, wave = threadIdx.x >> 5;
  __shared__ float gh[96];
  __shared__ __align__(16) float sh_h[1024];
  for (int s = 0; s < T; ++s) {
    const float* hold = hbase + (s & 1) * H;
    float*       hnew = hbase + ((s + 1) & 1) * H;
    const float* git  = gi + (size_t)(dir ? (T - 1 - s) : s) * G3H;
    // prefetch next step's gi rows while this step computes
    if (s < T - 1 && threadIdx.x < 3) {
      const float* gnext = gi + (size_t)(dir ? (T - 2 - s) : (s + 1)) * G3H;
      __builtin_prefetch(gnext + threadIdx.x * H + j0, 0, 0);
    }
    stage_h_to_lds(sh_h, hold);               // ASYNCcnt path
    gru_rows(Whh, bhh, sh_h, gh, j0, lane, wave);
    __syncthreads();
    if (threadIdx.x < 32) {
      int jj = threadIdx.x, j = j0 + jj;
      float gr = git[j]      + gh[jj];
      float gz = git[H + j]  + gh[32 + jj];
      float r  = 1.f / (1.f + __expf(-gr));
      float z  = 1.f / (1.f + __expf(-gz));
      float n  = tanhf(git[2 * H + j] + r * gh[64 + jj]);
      hnew[j]  = (1.f - z) * n + z * sh_h[j];
    }
    grid_sync(bar, 2 * NB);
  }
}

// Decoder GRU recurrence, 32 persistent blocks; gi fixed; hs emitted as bf16.
__global__ void __launch_bounds__(256, 1) dec_gru(
    const float* __restrict__ Whh, const float* __restrict__ bhh,
    const float* __restrict__ gid, float* __restrict__ hdec,
    __bf16* __restrict__ hsbf, int* bar) {
  const int NB = 32;
  int j0   = blockIdx.x * 32;
  int lane = threadIdx.x & 31, wave = threadIdx.x >> 5;
  __shared__ float gh[96];
  __shared__ __align__(16) float sh_h[1024];
  for (int s = 0; s < T; ++s) {
    const float* hold = hdec + (s & 1) * H;
    float*       hnew = hdec + ((s + 1) & 1) * H;
    stage_h_to_lds(sh_h, hold);               // ASYNCcnt path
    gru_rows(Whh, bhh, sh_h, gh, j0, lane, wave);
    __syncthreads();
    if (threadIdx.x < 32) {
      int jj = threadIdx.x, j = j0 + jj;
      float gr = gid[j]      + gh[jj];
      float gz = gid[H + j]  + gh[32 + jj];
      float r  = 1.f / (1.f + __expf(-gr));
      float z  = 1.f / (1.f + __expf(-gz));
      float n  = tanhf(gid[2 * H + j] + r * gh[64 + jj]);
      float v  = (1.f - z) * n + z * sh_h[j];
      hnew[j] = v;
      hsbf[(size_t)s * H + j] = (__bf16)v;
    }
    grid_sync(bar, NB);
  }
}

// ---------------------------------------------------------------------------
// Latent: mu/logvar/z  (u = [hf, hb], 256 outputs x 2048 MACs each)
// ---------------------------------------------------------------------------
__global__ void latent_z(const float* __restrict__ henc,
                         const float* __restrict__ zmuW, const float* __restrict__ zmub,
                         const float* __restrict__ zlvW, const float* __restrict__ zlvb,
                         const float* __restrict__ eps,
                         float* __restrict__ zout,
                         float* __restrict__ out_mu, float* __restrict__ out_lv) {
  int j = threadIdx.x;                        // 256
  const float* hf = henc;                     // final fwd h (buffer 0)
  const float* hb = henc + 2048;              // final bwd h (buffer 0)
  const float* wm = zmuW + (size_t)j * (2 * H);
  const float* wl = zlvW + (size_t)j * (2 * H);
  float mu = zmub[j], lv = zlvb[j];
  for (int k = 0; k < H; ++k) { float u = hf[k]; mu += u * wm[k];     lv += u * wl[k]; }
  for (int k = 0; k < H; ++k) { float u = hb[k]; mu += u * wm[H + k]; lv += u * wl[H + k]; }
  mu = fmaxf(mu, 0.f);
  lv = fmaxf(lv, 0.f);
  out_mu[j] = mu;
  out_lv[j] = lv;
  zout[j] = mu + eps[j] * __expf(0.5f * lv);
}

__global__ void dec_in_proj(const float* __restrict__ zin, const float* __restrict__ y_t,
                            const float* __restrict__ decWih,
                            const float* __restrict__ decbih,
                            float* __restrict__ gid) {
  int row = blockIdx.x * 256 + threadIdx.x;   // 3072
  const float* w = decWih + (size_t)row * (Z + LAB);
  float a = decbih[row];
  for (int k = 0; k < Z; ++k)   a += zin[k] * w[k];
  for (int k = 0; k < LAB; ++k) a += y_t[k] * w[Z + k];
  gid[row] = a;
}

// ---------------------------------------------------------------------------
// Host side
// ---------------------------------------------------------------------------
static inline size_t alignup(size_t x) { return (x + 255) & ~(size_t)255; }

extern "C" void kernel_launch(void* const* d_in, const int* in_sizes, int n_in,
                              void* d_out, int out_size, void* d_ws, size_t ws_size,
                              hipStream_t stream) {
  const float* x_s     = (const float*)d_in[0];
  const float* y_t     = (const float*)d_in[1];
  const float* eps     = (const float*)d_in[2];
  const float* fwd_Wih = (const float*)d_in[3];
  const float* fwd_Whh = (const float*)d_in[4];
  const float* fwd_bih = (const float*)d_in[5];
  const float* fwd_bhh = (const float*)d_in[6];
  const float* bwd_Wih = (const float*)d_in[7];
  const float* bwd_Whh = (const float*)d_in[8];
  const float* bwd_bih = (const float*)d_in[9];
  const float* bwd_bhh = (const float*)d_in[10];
  const float* zmu_W   = (const float*)d_in[11];
  const float* zmu_b   = (const float*)d_in[12];
  const float* zlv_W   = (const float*)d_in[13];
  const float* zlv_b   = (const float*)d_in[14];
  const float* dec_Wih = (const float*)d_in[15];
  const float* dec_Whh = (const float*)d_in[16];
  const float* dec_bih = (const float*)d_in[17];
  const float* dec_bhh = (const float*)d_in[18];
  const float* fc_W    = (const float*)d_in[19];
  const float* fc_b    = (const float*)d_in[20];

  float* out = (float*)d_out;
  float* out_xt = out;                        // 512*10000
  float* out_mu = out + (size_t)T * V;        // 256
  float* out_lv = out_mu + Z;                 // 256

  char* ws = (char*)d_ws;
  size_t off = 0;
  __bf16* xbf  = (__bf16*)(ws + off); off = alignup(off + (size_t)T * VP * 2);
  __bf16* wfbf = (__bf16*)(ws + off); off = alignup(off + (size_t)G3H * VP * 2);
  __bf16* wbbf = (__bf16*)(ws + off); off = alignup(off + (size_t)G3H * VP * 2);
  __bf16* fcbf = (__bf16*)(ws + off); off = alignup(off + (size_t)NPAD * H * 2);
  float*  gif  = (float*)(ws + off);  off = alignup(off + (size_t)T * G3H * 4);
  float*  gib  = (float*)(ws + off);  off = alignup(off + (size_t)T * G3H * 4);
  float*  henc = (float*)(ws + off);  off = alignup(off + 4096 * 4);
  float*  hdec = (float*)(ws + off);  off = alignup(off + 2048 * 4);
  float*  zbuf = (float*)(ws + off);  off = alignup(off + Z * 4);
  float*  gid  = (float*)(ws + off);  off = alignup(off + G3H * 4);
  __bf16* hsbf = (__bf16*)(ws + off); off = alignup(off + (size_t)T * H * 2);
  int*    bar  = (int*)(ws + off);    off = alignup(off + 256);
  (void)ws_size; (void)in_sizes; (void)n_in; (void)out_size;

  init_state<<<1, 256, 0, stream>>>(henc, hdec, bar);

  // bf16 staging: 2D, division-free, coalesced
  cvt_pad_cols<<<dim3(40, T),    256, 0, stream>>>(x_s,     xbf,  V, VP);
  cvt_pad_cols<<<dim3(40, G3H),  256, 0, stream>>>(fwd_Wih, wfbf, V, VP);
  cvt_pad_cols<<<dim3(40, G3H),  256, 0, stream>>>(bwd_Wih, wbbf, V, VP);
  cvt_pad_rows<<<dim3(4, NPAD),  256, 0, stream>>>(fc_W,    fcbf, V, H);

  gemm_enc_proj<<<384, 256, 0, stream>>>(xbf, wfbf, wbbf, fwd_bih, bwd_bih,
                                         gif, gib);

  enc_gru<<<64, 256, 0, stream>>>(fwd_Whh, fwd_bhh, bwd_Whh, bwd_bhh,
                                  gif, gib, henc, bar);

  latent_z<<<1, 256, 0, stream>>>(henc, zmu_W, zmu_b, zlv_W, zlv_b, eps,
                                  zbuf, out_mu, out_lv);
  dec_in_proj<<<12, 256, 0, stream>>>(zbuf, y_t, dec_Wih, dec_bih, gid);

  dec_gru<<<32, 256, 0, stream>>>(dec_Whh, dec_bhh, gid, hdec, hsbf, bar);

  gemm_out_proj<<<626, 256, 0, stream>>>(hsbf, fcbf, fc_b, out_xt);
}
